// DFVAE_67826123538573
// MI455X (gfx1250) — compile-verified, matrix operand order checked by reading, not run on hardware
//
#include <hip/hip_runtime.h>

#define NROWS 16384
#define LD    512
#define BM    128
#define BN    128
#define BK    64
#define AST   72            // padded LDS row stride in bf16 elems (144B, 16B-aligned, conflict-free)
#define NKT   (LD / BK)     // 8 K-tiles

#if defined(__has_builtin)
#if __has_builtin(__builtin_amdgcn_global_load_async_to_lds_b128) && __has_builtin(__builtin_amdgcn_s_wait_asynccnt)
#define USE_ASYNC 1
#endif
#endif
#ifndef USE_ASYNC
#define USE_ASYNC 0
#endif

typedef __attribute__((ext_vector_type(16))) __bf16         v16bf;
typedef __attribute__((ext_vector_type(8)))  float          v8f;
typedef __attribute__((ext_vector_type(4)))  unsigned int   v4u;
typedef __attribute__((ext_vector_type(4)))  int            v4i;
typedef __attribute__((ext_vector_type(4)))  unsigned short v4us;

union Frag { v16bf bf; v4u u[2]; };

__device__ __forceinline__ unsigned short f2bf(float f) {
  unsigned int u = __builtin_bit_cast(unsigned int, f);
  u += 0x7FFFu + ((u >> 16) & 1u);                 // round-to-nearest-even
  return (unsigned short)(u >> 16);
}

__device__ __forceinline__ unsigned int f2bf2(float a, float b) {
#if defined(__AMDGCN__) && __has_builtin(__builtin_amdgcn_cvt_pk_bf16_f32)
  typedef __attribute__((ext_vector_type(2))) __bf16 v2bf;
  v2bf r = __builtin_amdgcn_cvt_pk_bf16_f32(a, b);
  return __builtin_bit_cast(unsigned int, r);
#else
  return (unsigned int)f2bf(a) | ((unsigned int)f2bf(b) << 16);
#endif
}

// 16-byte global(bf16) -> LDS copy; async (ASYNCcnt) when available.
__device__ __forceinline__ void copy16(const unsigned short* g, unsigned short* l) {
#if USE_ASYNC
  __builtin_amdgcn_global_load_async_to_lds_b128(
      (__attribute__((address_space(1))) v4i*)(unsigned long long)g,
      (__attribute__((address_space(3))) v4i*)(unsigned int)(unsigned long long)l,
      0, 0);
#else
  *(v4u*)l = *(const v4u*)g;
#endif
}

template<int N>
__device__ __forceinline__ void wait_async() {
#if USE_ASYNC
  __builtin_amdgcn_s_wait_asynccnt(N);
#endif
}

// Stage A tile (BM x BK). bf16 input: 4x16B copies/thread. fp32 input: load+convert.
template<bool IN_BF16>
__device__ __forceinline__ void issue_A(const void* zin, int blockRow, int k0,
                                        unsigned short* buf, int tid) {
  if (IN_BF16) {
    const unsigned short* z = (const unsigned short*)zin;
    const int kc = tid & 7, r0 = tid >> 3;          // 8 k-chunks x 32 rows
    #pragma unroll
    for (int i = 0; i < 4; ++i) {
      const int r = r0 + 32 * i;
      copy16(z + (size_t)(blockRow + r) * LD + k0 + kc * 8, buf + r * AST + kc * 8);
    }
  } else {
    const float* z = (const float*)zin;
    const int kg = tid & 15, r0 = tid >> 4;         // 16 float4-groups x 16 rows
    #pragma unroll
    for (int i = 0; i < 8; ++i) {
      const int r = r0 + 16 * i;
      const float4 v = *(const float4*)(z + (size_t)(blockRow + r) * LD + k0 + kg * 4);
      const unsigned long long lo = f2bf2(v.x, v.y);
      const unsigned long long hi = f2bf2(v.z, v.w);
      *(unsigned long long*)(buf + r * AST + kg * 4) = lo | (hi << 32);
    }
  }
}

// Stage B tile (BN x BK) from pre-transposed bf16 weights Wt[n][k].
__device__ __forceinline__ void issue_B(const unsigned short* Wte, int blockCol, int k0,
                                        unsigned short* buf, int tid) {
  const int kc = tid & 7, n0 = tid >> 3;
  #pragma unroll
  for (int i = 0; i < 4; ++i) {
    const int n = n0 + 32 * i;
    copy16(Wte + (size_t)(blockCol + n) * LD + k0 + kc * 8, buf + n * AST + kc * 8);
  }
}

template<bool MASKED>
__device__ __forceinline__ void compute_tile(const unsigned short* As, const unsigned short* Bs,
                                             v8f acc[2][4], bool act0, bool act1,
                                             int lane16, int half, int wm, int wn) {
  #pragma unroll
  for (int kk = 0; kk < BK; kk += 32) {
    Frag aF[2];
    #pragma unroll
    for (int mi = 0; mi < 2; ++mi) {
      const int row = 32 * wm + 16 * mi + lane16;
      aF[mi].u[0] = *(const v4u*)&As[row * AST + kk + 8 * half];       // K 0-7 | 8-15
      aF[mi].u[1] = *(const v4u*)&As[row * AST + kk + 16 + 8 * half];  // K 16-23 | 24-31
      if (MASKED) {
        const bool act = mi ? act1 : act0;
        aF[mi].u[0] = act ? aF[mi].u[0] : (v4u)0u;
        aF[mi].u[1] = act ? aF[mi].u[1] : (v4u)0u;
      }
    }
    Frag bF[4];
    #pragma unroll
    for (int ni = 0; ni < 4; ++ni) {
      const int col = 64 * wn + 16 * ni + lane16;
      bF[ni].u[0] = *(const v4u*)&Bs[col * AST + kk + 8 * half];
      bF[ni].u[1] = *(const v4u*)&Bs[col * AST + kk + 16 + 8 * half];
    }
    #pragma unroll
    for (int mi = 0; mi < 2; ++mi)
      #pragma unroll
      for (int ni = 0; ni < 4; ++ni)
        acc[mi][ni] = __builtin_amdgcn_wmma_f32_16x16x32_bf16(
            false, aF[mi].bf, false, bF[ni].bf, (short)0, acc[mi][ni], false, false);
  }
}

// Full-K double-buffered pipeline for one expert.
template<bool IN_BF16, bool MASKED>
__device__ __forceinline__ void run_expert(const void* zin, const unsigned short* Wte,
                                           int blockRow, int blockCol, int tid,
                                           v8f acc[2][4], unsigned short* As, unsigned short* Bs,
                                           bool act0, bool act1,
                                           int lane16, int half, int wm, int wn) {
  unsigned short* Ac = As;            unsigned short* An = As + BM * AST;
  unsigned short* Bc = Bs;            unsigned short* Bn = Bs + BN * AST;
  issue_A<IN_BF16>(zin, blockRow, 0, Ac, tid);
  issue_B(Wte, blockCol, 0, Bc, tid);
  #pragma unroll 2
  for (int t = 0; t < NKT; ++t) {
    if (t + 1 < NKT) {
      issue_A<IN_BF16>(zin, blockRow, (t + 1) * BK, An, tid);
      issue_B(Wte, blockCol, (t + 1) * BK, Bn, tid);
      wait_async<(IN_BF16 ? 8 : 4)>();   // in-order completion: tile t fully landed
    } else {
      wait_async<0>();
    }
    __syncthreads();
    compute_tile<MASKED>(Ac, Bc, acc, act0, act1, lane16, half, wm, wn);
    __syncthreads();
    unsigned short* tA = Ac; Ac = An; An = tA;
    unsigned short* tB = Bc; Bc = Bn; Bn = tB;
  }
}

template<bool IN_BF16, bool OUT_BF16>
__global__ __launch_bounds__(256)
void moe_stage(const void* __restrict__ zin, const unsigned short* __restrict__ Wt,
               const float* __restrict__ bias, const int* __restrict__ ids,
               void* __restrict__ zout) {
  __shared__ unsigned short As[2 * BM * AST];
  __shared__ unsigned short Bs[2 * BN * AST];

  const int tid      = threadIdx.x;
  const int blockRow = blockIdx.x * BM;
  const int blockCol = blockIdx.y * BN;
  const int wave   = tid >> 5;
  const int lane   = tid & 31;
  const int lane16 = lane & 15;
  const int half   = lane >> 4;
  const int wm = wave & 3;
  const int wn = wave >> 2;

  v8f acc[2][4];
  #pragma unroll
  for (int i = 0; i < 2; ++i)
    #pragma unroll
    for (int j = 0; j < 4; ++j) acc[i][j] = (v8f)0.0f;

  const int eFirst = __builtin_amdgcn_readfirstlane(ids[blockRow]);
  const int eLast  = __builtin_amdgcn_readfirstlane(ids[blockRow + BM - 1]);

  if (eFirst == eLast) {
    // uniform block (common case): no masking in the WMMA loop
    run_expert<IN_BF16, false>(zin, Wt + (size_t)eFirst * LD * LD, blockRow, blockCol,
                               tid, acc, As, Bs, true, true, lane16, half, wm, wn);
  } else {
    for (int e = eFirst; e <= eLast; ++e) {
      const bool a0 = (ids[blockRow + 32 * wm + lane16] == e);
      const bool a1 = (ids[blockRow + 32 * wm + 16 + lane16] == e);
      run_expert<IN_BF16, true>(zin, Wt + (size_t)e * LD * LD, blockRow, blockCol,
                                tid, acc, As, Bs, a0, a1, lane16, half, wm, wn);
    }
  }

  // epilogue: per-row expert bias + ReLU; C/D layout: VGPR r -> M=r+8*half, lane16 -> N
  #pragma unroll
  for (int mi = 0; mi < 2; ++mi) {
    const int rowBase = blockRow + 32 * wm + 16 * mi + 8 * half;
    #pragma unroll
    for (int ni = 0; ni < 4; ++ni) {
      const int col = blockCol + 64 * wn + 16 * ni + lane16;
      #pragma unroll
      for (int r = 0; r < 8; ++r) {
        const int row = rowBase + r;
        const int er  = ids[row];
        float v = acc[mi][ni][r] + bias[er * LD + col];
        v = v > 0.0f ? v : 0.0f;
        if (OUT_BF16) ((unsigned short*)zout)[(size_t)row * LD + col] = f2bf(v);
        else          ((float*)zout)[(size_t)row * LD + col] = v;
      }
    }
  }
}

// One-time: convert expert weights fp32 [e][k][n] -> bf16 transposed [e][n][k].
__global__ __launch_bounds__(256)
void wcvt(const float* __restrict__ W, unsigned short* __restrict__ Wt) {
  __shared__ float t[32][33];
  const int e  = blockIdx.z;
  const int n0 = blockIdx.x * 32, k0 = blockIdx.y * 32;
  const int tx = threadIdx.x & 31, ty = threadIdx.x >> 5;   // 32 x 8
  const float* We = W + (size_t)e * LD * LD;
  unsigned short* Wte = Wt + (size_t)e * LD * LD;
  #pragma unroll
  for (int i = 0; i < 4; ++i) {
    const int k = ty + 8 * i;
    t[k][tx] = We[(size_t)(k0 + k) * LD + n0 + tx];          // coalesced along n
  }
  __syncthreads();
  #pragma unroll
  for (int i = 0; i < 4; ++i) {
    const int n = ty + 8 * i;
    Wte[(size_t)(n0 + n) * LD + k0 + tx] = f2bf(t[tx][n]);   // coalesced along k
  }
}

extern "C" void kernel_launch(void* const* d_in, const int* in_sizes, int n_in,
                              void* d_out, int out_size, void* d_ws, size_t ws_size,
                              hipStream_t stream) {
  const float* z  = (const float*)d_in[0];
  const float* Wd = (const float*)d_in[1];
  const float* bd = (const float*)d_in[2];
  const float* Wa = (const float*)d_in[3];
  const float* ba = (const float*)d_in[4];
  const float* Wn = (const float*)d_in[5];
  const float* bn = (const float*)d_in[6];
  const int* idd  = (const int*)d_in[7];
  const int* ida  = (const int*)d_in[8];
  const int* idn  = (const int*)d_in[9];

  char* ws = (char*)d_ws;
  const size_t WSZ = (size_t)LD * LD * 2;                 // bytes per bf16 expert matrix
  unsigned short* WtD = (unsigned short*)(ws);            // 8 experts
  unsigned short* WtA = (unsigned short*)(ws + 8  * WSZ); // 6 experts
  unsigned short* WtN = (unsigned short*)(ws + 14 * WSZ); // 16 experts
  unsigned short* z2  = (unsigned short*)(ws + 30 * WSZ); // stage-2 bf16 output (16 MB)
  unsigned short* z1  = (unsigned short*)d_out;           // stage-1 bf16 output aliases d_out

  dim3 cb(256);
  wcvt<<<dim3(LD / 32, LD / 32, 8),  cb, 0, stream>>>(Wd, WtD);
  wcvt<<<dim3(LD / 32, LD / 32, 6),  cb, 0, stream>>>(Wa, WtA);
  wcvt<<<dim3(LD / 32, LD / 32, 16), cb, 0, stream>>>(Wn, WtN);

  dim3 grid(NROWS / BM, LD / BN), block(256);
  moe_stage<false, true ><<<grid, block, 0, stream>>>(z,  WtD, bd, idd, z1);
  moe_stage<true,  true ><<<grid, block, 0, stream>>>(z1, WtA, ba, ida, z2);
  moe_stage<true,  false><<<grid, block, 0, stream>>>(z2, WtN, bn, idn, d_out);
}
